// AttentionBlock_15341623181566
// MI455X (gfx1250) — matrix operand order repaired and checked by reading, not compiled, not run on hardware
//
#include <hip/hip_runtime.h>

// ---------------------------------------------------------------------------
// AttentionBlock for MI455X (gfx1250, wave32, WMMA).
// b=16, c=512, h=w=32 (n=1024), heads=4, head_dim=128, groups=32.
// All GEMMs via v_wmma_f32_16x16x32_f16 (f16 in, f32 accumulate).
// GEMM waves own 32x64 output tiles (2Mx4N) for fragment reuse.
// ---------------------------------------------------------------------------

typedef _Float16 v16h __attribute__((ext_vector_type(16)));
typedef _Float16 v8h  __attribute__((ext_vector_type(8)));
typedef float    v8f  __attribute__((ext_vector_type(8)));

#define BATCH   16
#define CCH     512
#define NTOK    1024
#define NHEADS  4
#define HD      128
#define NGROUPS 32
#define Q_SCALE 0.08838834764831845f   // 1/sqrt(128)

__device__ __forceinline__ v8f wmma_f32(v16h a, v16h b, v8f c) {
  // (neg_a, A, neg_b, B, c_mod, C, reuse_a, reuse_b)
  return __builtin_amdgcn_wmma_f32_16x16x32_f16(false, a, false, b, (short)0, c,
                                                false, false);
}

// A-fragment (16x32 f16, M x K). Row M = lane%16 for all lanes.
// half 0 (lanes 0-15): K = {k0+0..15}; half 1: {k0+16..31}.
// Caller folds (sub*16) into p; we load p[0..7] and p[8..15].
__device__ __forceinline__ v16h make_a(const _Float16* p) {
  v8h lo = *(const v8h*)p;
  v8h hi = *(const v8h*)(p + 8);
  v16h a;
#pragma unroll
  for (int i = 0; i < 8; ++i) { a[i] = lo[i]; a[i + 8] = hi[i]; }
  return a;
}

__device__ __forceinline__ float rmax16(float v) {
#pragma unroll
  for (int m = 1; m < 16; m <<= 1) v = fmaxf(v, __shfl_xor(v, m, 32));
  return v;
}
__device__ __forceinline__ float rsum16(float v) {
#pragma unroll
  for (int m = 1; m < 16; m <<= 1) v += __shfl_xor(v, m, 32);
  return v;
}

// ---------------------------------------------------------------------------
// Kernel 1: fp32 -> f16 weight conversion.
// ---------------------------------------------------------------------------
__global__ void cvt_w_kernel(const float* __restrict__ qkv_w,
                             const float* __restrict__ proj_w,
                             _Float16* __restrict__ wq,
                             _Float16* __restrict__ wp) {
  int i = blockIdx.x * 256 + threadIdx.x;
  if (i < 3 * CCH * CCH) wq[i] = (_Float16)qkv_w[i];
  if (i < CCH * CCH)     wp[i] = (_Float16)proj_w[i];
}

// ---------------------------------------------------------------------------
// Kernel 2: GroupNorm -> Yt[b][t][c] (token-major f16).
// One block per (batch, group): 16 channels x 1024 tokens.
// ---------------------------------------------------------------------------
__global__ __launch_bounds__(256) void gn_kernel(const float* __restrict__ x,
                                                 const float* __restrict__ nw,
                                                 const float* __restrict__ nb,
                                                 _Float16* __restrict__ Yt) {
  const int b = blockIdx.x >> 5;
  const int g = blockIdx.x & 31;
  const float* xp = x + ((size_t)b * CCH + g * 16) * NTOK;

  float s = 0.f, s2 = 0.f;
  for (int i = threadIdx.x; i < 16 * NTOK; i += 256) {
    float v = xp[i];
    s += v; s2 += v * v;
  }
#pragma unroll
  for (int m = 1; m < 32; m <<= 1) { s += __shfl_xor(s, m, 32); s2 += __shfl_xor(s2, m, 32); }

  __shared__ float red[2][8];
  const int wid = threadIdx.x >> 5, lane = threadIdx.x & 31;
  if (lane == 0) { red[0][wid] = s; red[1][wid] = s2; }
  __syncthreads();
  s = 0.f; s2 = 0.f;
#pragma unroll
  for (int j = 0; j < 8; ++j) { s += red[0][j]; s2 += red[1][j]; }

  const float mean = s * (1.f / (16.f * NTOK));
  const float var  = s2 * (1.f / (16.f * NTOK)) - mean * mean;
  const float inv  = rsqrtf(var + 1e-5f);

  float sc[16], sh[16];
#pragma unroll
  for (int ci = 0; ci < 16; ++ci) {
    const int c = g * 16 + ci;
    sc[ci] = nw[c] * inv;
    sh[ci] = nb[c] - mean * sc[ci];
  }
  for (int t = threadIdx.x; t < NTOK; t += 256) {
    v16h o;
#pragma unroll
    for (int ci = 0; ci < 16; ++ci)
      o[ci] = (_Float16)(xp[ci * NTOK + t] * sc[ci] + sh[ci]);
    *(v16h*)(Yt + ((size_t)b * NTOK + t) * CCH + g * 16) = o;
  }
}

// ---------------------------------------------------------------------------
// Kernel 3: QKV GEMM (1536 x 1024, K=512 per batch) via WMMA.
// Each wave: 32x64 output tile (2 M-tiles x 4 N-tiles), A reused across N,
// B reused across M -> 8 WMMAs per (2 A + 4 B) fragment loads per K-chunk.
// Writes Qt/Kt[b][h][t][d] (token-major, Q pre-scaled) and Vt[b][h][d][t].
// ---------------------------------------------------------------------------
__global__ __launch_bounds__(128) void qkv_kernel(const _Float16* __restrict__ Wq,
                                                  const float* __restrict__ qkv_b,
                                                  const _Float16* __restrict__ Yt,
                                                  _Float16* __restrict__ Qt,
                                                  _Float16* __restrict__ Kt,
                                                  _Float16* __restrict__ Vt) {
  const int wid = threadIdx.x >> 5, lane = threadIdx.x & 31;
  const int sub = lane >> 4, lh = lane & 15;

  int tile = blockIdx.x * 4 + wid;          // 12288 waves total
  const int nt = tile & 15;  tile >>= 4;    // 16 N-tiles of 64 tokens
  const int mt = tile % 48;                 // 48 M-tiles of 32 outputs
  const int b  = tile / 48;
  const int t0 = nt * 64, o0 = mt * 32;

  const _Float16* arow = Wq + (size_t)(o0 + lh) * CCH + sub * 16;
  const _Float16* brow = Yt + ((size_t)b * NTOK + t0 + lh) * CCH + sub * 16;

  v8f acc[2][4];
#pragma unroll
  for (int mi = 0; mi < 2; ++mi)
#pragma unroll
    for (int j = 0; j < 4; ++j)
      acc[mi][j] = v8f{0.f, 0.f, 0.f, 0.f, 0.f, 0.f, 0.f, 0.f};

  for (int k0 = 0; k0 < CCH; k0 += 32) {
    __builtin_prefetch(arow + k0 + 32, 0, 3);   // WGP-scope global_prefetch_b8
    __builtin_prefetch(brow + k0 + 32, 0, 3);
    const v16h a0 = make_a(arow + k0);
    const v16h a1 = make_a(arow + 16 * CCH + k0);     // +8192 elems: imm offset
#pragma unroll
    for (int j = 0; j < 4; ++j) {
      const v16h bf = *(const v16h*)(brow + j * 16 * CCH + k0);
      acc[0][j] = wmma_f32(a0, bf, acc[0][j]);
      acc[1][j] = wmma_f32(a1, bf, acc[1][j]);
    }
  }

#pragma unroll
  for (int mi = 0; mi < 2; ++mi) {
    const int o0m   = o0 + mi * 16;
    const int which = o0m >> 9;          // 0:q 1:k 2:v
    const int oc    = o0m & 511;
    const int head  = oc >> 7;
    const int dbase = (oc & 127) + sub * 8;
#pragma unroll
    for (int j = 0; j < 4; ++j) {
      const int t0j = t0 + j * 16;
      const size_t tok = (size_t)(b * NHEADS + head) * NTOK + t0j + lh;
      if (which == 0) {
        v8h o;
#pragma unroll
        for (int e = 0; e < 8; ++e)
          o[e] = (_Float16)((acc[mi][j][e] + qkv_b[o0m + sub * 8 + e]) * Q_SCALE);
        *(v8h*)(Qt + tok * HD + dbase) = o;
      } else if (which == 1) {
        v8h o;
#pragma unroll
        for (int e = 0; e < 8; ++e)
          o[e] = (_Float16)(acc[mi][j][e] + qkv_b[o0m + sub * 8 + e]);
        *(v8h*)(Kt + tok * HD + dbase) = o;
      } else {
#pragma unroll
        for (int e = 0; e < 8; ++e)
          Vt[((size_t)(b * NHEADS + head) * HD + dbase + e) * NTOK + t0j + lh] =
              (_Float16)(acc[mi][j][e] + qkv_b[o0m + sub * 8 + e]);
      }
    }
  }
}

// ---------------------------------------------------------------------------
// Kernel 4: fused flash attention. Block = 8 waves = 128 queries of one
// (batch, head); wave = 16 queries; key blocks of 32.
// S = Q*K^T (C layout: M=query, N=key); online softmax with 16-lane shuffle
// reductions; P bounced through LDS; O' = V * P^T so the P B-fragment is one
// contiguous 32B LDS read and output d is contiguous per lane.
// ---------------------------------------------------------------------------
__global__ __launch_bounds__(256) void attn_kernel(const _Float16* __restrict__ Qt,
                                                   const _Float16* __restrict__ Kt,
                                                   const _Float16* __restrict__ Vt,
                                                   _Float16* __restrict__ Ot) {
  const int wid = threadIdx.x >> 5, lane = threadIdx.x & 31;
  const int sub = lane >> 4, lh = lane & 15;
  const int head = blockIdx.y, b = blockIdx.z;
  const int bh = b * NHEADS + head;
  const int qbase = blockIdx.x * 128 + wid * 16;

  const _Float16* Qp = Qt + ((size_t)bh * NTOK + qbase) * HD;
  const _Float16* Kp = Kt + (size_t)bh * NTOK * HD;
  const _Float16* Vp = Vt + (size_t)bh * HD * NTOK;

  __shared__ _Float16 Pbuf[8][16][32];
  __shared__ float    Sc[8][16];

  // Q A-fragments, K chunks of 32 over head_dim=128.
  v16h qa[4];
#pragma unroll
  for (int kc = 0; kc < 4; ++kc)
    qa[kc] = make_a(Qp + (size_t)lh * HD + kc * 32 + sub * 16);

  v8f o_acc[8];
#pragma unroll
  for (int dt = 0; dt < 8; ++dt)
    o_acc[dt] = v8f{0.f, 0.f, 0.f, 0.f, 0.f, 0.f, 0.f, 0.f};

  float m_st[8], l_st[8];
#pragma unroll
  for (int r = 0; r < 8; ++r) { m_st[r] = -1e30f; l_st[r] = 0.f; }

  for (int m0 = 0; m0 < NTOK; m0 += 32) {
    v8f s0 = {0.f, 0.f, 0.f, 0.f, 0.f, 0.f, 0.f, 0.f};
    v8f s1 = {0.f, 0.f, 0.f, 0.f, 0.f, 0.f, 0.f, 0.f};
#pragma unroll
    for (int kc = 0; kc < 4; ++kc) {
      v16h b0 = *(const v16h*)(Kp + (size_t)(m0 + lh) * HD + kc * 32 + sub * 16);
      v16h b1 = *(const v16h*)(Kp + (size_t)(m0 + 16 + lh) * HD + kc * 32 + sub * 16);
      s0 = wmma_f32(qa[kc], b0, s0);
      s1 = wmma_f32(qa[kc], b1, s1);
    }

    // Online softmax. C row r (+8*sub) spans the 16 lanes of this half-wave.
    float alpha[8];
#pragma unroll
    for (int r = 0; r < 8; ++r) {
      const float rowmax = rmax16(fmaxf(s0[r], s1[r]));
      const float mnew   = fmaxf(m_st[r], rowmax);
      const float p0 = __expf(s0[r] - mnew);
      const float p1 = __expf(s1[r] - mnew);
      const float rowsum = rsum16(p0 + p1);
      alpha[r] = __expf(m_st[r] - mnew);
      l_st[r]  = l_st[r] * alpha[r] + rowsum;
      m_st[r]  = mnew;
      Pbuf[wid][r + 8 * sub][lh]      = (_Float16)p0;
      Pbuf[wid][r + 8 * sub][16 + lh] = (_Float16)p1;
    }
    // Broadcast per-query rescale factors (values are half-wave uniform).
    Sc[wid][sub * 8 + (lh & 7)] = alpha[lh & 7];
    asm volatile("s_wait_dscnt 0" ::: "memory");
    const float al = Sc[wid][lh];

#pragma unroll
    for (int dt = 0; dt < 8; ++dt) {
      o_acc[dt] = o_acc[dt] * al;  // this lane's column is query lh
      // A = V rows d (M = dt*16 + lh), K = keys m0..m0+31.
      v16h va = make_a(Vp + (size_t)(dt * 16 + lh) * NTOK + m0 + sub * 16);
      // B = P^T: lane column = query lh, K = keys contiguous in the LDS row.
      v16h pb = *(const v16h*)&Pbuf[wid][lh][sub * 16];
      o_acc[dt] = wmma_f32(va, pb, o_acc[dt]);
    }
  }

  // Final 1/l normalization, broadcast through LDS, then packed f16 stores.
  Sc[wid][sub * 8 + (lh & 7)] = 1.f / l_st[lh & 7];
  asm volatile("s_wait_dscnt 0" ::: "memory");
  const float invl = Sc[wid][lh];

  const size_t obase = ((size_t)b * NTOK + qbase + lh) * CCH + head * HD;
#pragma unroll
  for (int dt = 0; dt < 8; ++dt) {
    v8h o;
#pragma unroll
    for (int e = 0; e < 8; ++e) o[e] = (_Float16)(o_acc[dt][e] * invl);
    *(v8h*)(Ot + obase + dt * 16 + sub * 8) = o;
  }
}

// ---------------------------------------------------------------------------
// Kernel 5: proj GEMM (512 x 1024, K=512) + bias + residual, fp32 output.
// Same 32x64-per-wave tiling as the QKV GEMM.
// ---------------------------------------------------------------------------
__global__ __launch_bounds__(128) void proj_kernel(const _Float16* __restrict__ Wp,
                                                   const float* __restrict__ proj_b,
                                                   const _Float16* __restrict__ Ot,
                                                   const float* __restrict__ x,
                                                   float* __restrict__ out) {
  const int wid = threadIdx.x >> 5, lane = threadIdx.x & 31;
  const int sub = lane >> 4, lh = lane & 15;

  int tile = blockIdx.x * 4 + wid;          // 4096 waves total
  const int nt = tile & 15;  tile >>= 4;    // 16 N-tiles of 64 tokens
  const int mt = tile & 15;                 // 16 M-tiles of 32 outputs
  const int b  = tile >> 4;
  const int t0 = nt * 64, o0 = mt * 32;

  const _Float16* arow = Wp + (size_t)(o0 + lh) * CCH + sub * 16;
  const _Float16* brow = Ot + ((size_t)b * NTOK + t0 + lh) * CCH + sub * 16;

  v8f acc[2][4];
#pragma unroll
  for (int mi = 0; mi < 2; ++mi)
#pragma unroll
    for (int j = 0; j < 4; ++j)
      acc[mi][j] = v8f{0.f, 0.f, 0.f, 0.f, 0.f, 0.f, 0.f, 0.f};

  for (int k0 = 0; k0 < CCH; k0 += 32) {
    __builtin_prefetch(arow + k0 + 32, 0, 3);
    __builtin_prefetch(brow + k0 + 32, 0, 3);
    const v16h a0 = make_a(arow + k0);
    const v16h a1 = make_a(arow + 16 * CCH + k0);
#pragma unroll
    for (int j = 0; j < 4; ++j) {
      const v16h bf = *(const v16h*)(brow + j * 16 * CCH + k0);
      acc[0][j] = wmma_f32(a0, bf, acc[0][j]);
      acc[1][j] = wmma_f32(a1, bf, acc[1][j]);
    }
  }

#pragma unroll
  for (int mi = 0; mi < 2; ++mi)
#pragma unroll
    for (int j = 0; j < 4; ++j)
#pragma unroll
      for (int e = 0; e < 8; ++e) {
        const int o = o0 + mi * 16 + sub * 8 + e;
        const size_t idx = ((size_t)b * CCH + o) * NTOK + t0 + j * 16 + lh;
        out[idx] = x[idx] + proj_b[o] + acc[mi][j][e];
      }
}

// ---------------------------------------------------------------------------
// Launch.  Workspace layout (bytes):
//   Wq f16  [0,            1,572,864)
//   Wp f16  [1,572,864,    2,097,152)
//   Yt f16  [2,097,152,   18,874,368)   (aliased as Ot after QKV has read it)
//   Qt f16  [18,874,368,  35,651,584)
//   Kt f16  [35,651,584,  52,428,800)
//   Vt f16  [52,428,800,  69,206,016)
// ---------------------------------------------------------------------------
extern "C" void kernel_launch(void* const* d_in, const int* in_sizes, int n_in,
                              void* d_out, int out_size, void* d_ws, size_t ws_size,
                              hipStream_t stream) {
  const float* x      = (const float*)d_in[0];
  const float* norm_w = (const float*)d_in[1];
  const float* norm_b = (const float*)d_in[2];
  const float* qkv_w  = (const float*)d_in[3];
  const float* qkv_b  = (const float*)d_in[4];
  const float* proj_w = (const float*)d_in[5];
  const float* proj_b = (const float*)d_in[6];
  float* out = (float*)d_out;

  char* ws = (char*)d_ws;
  _Float16* Wq = (_Float16*)(ws);
  _Float16* Wp = (_Float16*)(ws + 1572864);
  _Float16* Yt = (_Float16*)(ws + 2097152);
  _Float16* Ot = Yt;  // safe: attn (writer) runs after qkv (reader) on stream
  _Float16* Qt = (_Float16*)(ws + 18874368);
  _Float16* Kt = (_Float16*)(ws + 35651584);
  _Float16* Vt = (_Float16*)(ws + 52428800);

  cvt_w_kernel<<<3072, 256, 0, stream>>>(qkv_w, proj_w, Wq, Wp);
  gn_kernel<<<BATCH * NGROUPS, 256, 0, stream>>>(x, norm_w, norm_b, Yt);
  qkv_kernel<<<(BATCH * 48 * 16) / 4, 128, 0, stream>>>(Wq, qkv_b, Yt, Qt, Kt, Vt);
  attn_kernel<<<dim3(NTOK / 128, NHEADS, BATCH), 256, 0, stream>>>(Qt, Kt, Vt, Ot);
  proj_kernel<<<(BATCH * 16 * 16) / 4, 128, 0, stream>>>(Wp, proj_b, Ot, x, out);
}
